// MoEGenreGate_77919296684619
// MI455X (gfx1250) — matrix operand order repaired and checked by reading, not compiled, not run on hardware
//
#include <hip/hip_runtime.h>
#include <hip/hip_bf16.h>

typedef __attribute__((ext_vector_type(16))) _Float16 v16h;
typedef __attribute__((ext_vector_type(8)))  _Float16 v8h;
typedef __attribute__((ext_vector_type(8)))  float    v8f;

// Problem dims (fixed by setup_inputs)
constexpr int B_ = 2, S_ = 2048, H_ = 1024, G_ = 256, E_ = 8, M_ = 2048;
constexpr int NTOK   = B_ * S_;        // 4096 tokens
constexpr int CAP    = NTOK;           // per-expert token-list capacity
constexpr int TILE_T = 16;             // tokens per expert tile (WMMA M)
constexpr int TILES  = NTOK / TILE_T;  // 256 tiles per expert
constexpr float EPS_ = 1e-6f;

// ---- workspace layout (bytes, 256-aligned) ----
constexpr size_t WS_CNT = 0;                                    // E ints
constexpr size_t WS_GG  = 256;                                  // B*E floats
constexpr size_t WS_CW  = 512;                                  // NTOK*E floats (128KB)
constexpr size_t WS_TOK = WS_CW  + (size_t)NTOK * E_ * 4;       // E*CAP ints (128KB)
constexpr size_t WS_W1T = WS_TOK + (size_t)E_ * CAP * 4;        // f16 [E][M][H]  32MB
constexpr size_t WS_W2T = WS_W1T + (size_t)E_ * M_ * H_ * 2;    // f16 [E][M][M]  64MB
constexpr size_t WS_W3T = WS_W2T + (size_t)E_ * M_ * M_ * 2;    // f16 [E][H][M]  32MB

// LDS partition for expert kernel (dynamic)
constexpr int LDS_XB   = 0;                  // 16*1024 f16 = 32KB
constexpr int LDS_H1   = 32 * 1024;          // 16*2048 f16 = 64KB
constexpr int LDS_H2   = 96 * 1024;          // 16*2048 f16 = 64KB
constexpr int LDS_ROWW = 160 * 1024;         // 16 floats
constexpr int LDS_RTOK = 160 * 1024 + 64;    // 16 ints
constexpr int SMEM_BYTES = 160 * 1024 + 256;

// ---------------------------------------------------------------------------
// Kernel 1: transpose + f32->f16 convert.  src: [E][K][N] f32, dst: [E][N][K] f16
// ---------------------------------------------------------------------------
__global__ void transpose_cvt_kernel(const float* __restrict__ src,
                                     _Float16* __restrict__ dst,
                                     int K, int N) {
  __shared__ float tile[32][33];
  const size_t plane = (size_t)K * N;
  const float* s = src + (size_t)blockIdx.z * plane;
  _Float16*    d = dst + (size_t)blockIdx.z * plane;
  const int tx = threadIdx.x, ty = threadIdx.y;
  const int n0 = blockIdx.x * 32, k0 = blockIdx.y * 32;
#pragma unroll
  for (int j = 0; j < 32; j += 8)
    tile[ty + j][tx] = s[(size_t)(k0 + ty + j) * N + (n0 + tx)];
  __syncthreads();
#pragma unroll
  for (int j = 0; j < 32; j += 8)
    d[(size_t)(n0 + ty + j) * K + (k0 + tx)] = (_Float16)tile[tx][ty + j];
}

// ---------------------------------------------------------------------------
// Kernel 2: genre gate gg[b][e] = genre[b] . gg_W[:,e] + gg_b[e] + wg_b[e]
// ---------------------------------------------------------------------------
__global__ void genre_gate_kernel(const float* __restrict__ genre,
                                  const float* __restrict__ gg_W,
                                  const float* __restrict__ gg_b,
                                  const float* __restrict__ wg_b,
                                  float* __restrict__ gg) {
  const int b = blockIdx.x >> 3, e = blockIdx.x & 7;
  const int lane = threadIdx.x;
  float acc = 0.f;
  for (int g = lane; g < G_; g += 32)
    acc += genre[b * G_ + g] * gg_W[g * E_ + e];
#pragma unroll
  for (int m = 16; m > 0; m >>= 1) acc += __shfl_xor(acc, m, 32);
  if (lane == 0) gg[b * E_ + e] = acc + gg_b[e] + wg_b[e];
}

// ---------------------------------------------------------------------------
// Kernel 3: per-token gating, softmax, top-2, dispatch.  One wave per token.
// ---------------------------------------------------------------------------
__global__ __launch_bounds__(128)
void gate_topk_kernel(const float* __restrict__ x,
                      const float* __restrict__ rms_w,
                      const float* __restrict__ wg_W,
                      const float* __restrict__ gg,
                      float* __restrict__ cw,
                      int*   __restrict__ cnt,
                      int*   __restrict__ toklist) {
  const int wid  = threadIdx.x >> 5;
  const int lane = threadIdx.x & 31;
  const int token = blockIdx.x * 4 + wid;
  const float* xr = x + (size_t)token * H_;

  // cache this token's 32 strided elements per lane (coalesced b32 loads)
  float xl[32], rl[32];
#pragma unroll
  for (int i = 0; i < 32; i++) {
    xl[i] = xr[i * 32 + lane];
    rl[i] = rms_w[i * 32 + lane];
  }
  float ss = 0.f;
#pragma unroll
  for (int i = 0; i < 32; i++) ss += xl[i] * xl[i];
#pragma unroll
  for (int m = 16; m > 0; m >>= 1) ss += __shfl_xor(ss, m, 32);
  const float inv = 1.0f / sqrtf(ss * (1.0f / H_) + EPS_);

  float g[E_];
#pragma unroll
  for (int e = 0; e < E_; e++) g[e] = 0.f;
#pragma unroll
  for (int i = 0; i < 32; i++) {
    const float xn = xl[i] * rl[i];
    const float* wr = wg_W + (size_t)(i * 32 + lane) * E_;
    const float4 w0 = *(const float4*)(wr);
    const float4 w1 = *(const float4*)(wr + 4);
    g[0] += xn * w0.x; g[1] += xn * w0.y; g[2] += xn * w0.z; g[3] += xn * w0.w;
    g[4] += xn * w1.x; g[5] += xn * w1.y; g[6] += xn * w1.z; g[7] += xn * w1.w;
  }
#pragma unroll
  for (int e = 0; e < E_; e++) {
#pragma unroll
    for (int m = 16; m > 0; m >>= 1) g[e] += __shfl_xor(g[e], m, 32);
  }
  const int b = token >> 11;  // token / S_
#pragma unroll
  for (int e = 0; e < E_; e++) g[e] = g[e] * inv + gg[b * E_ + e];

  // softmax over 8 (redundant on all lanes)
  float mx = g[0];
#pragma unroll
  for (int e = 1; e < E_; e++) mx = fmaxf(mx, g[e]);
  float p[E_], sum = 0.f;
#pragma unroll
  for (int e = 0; e < E_; e++) { p[e] = expf(g[e] - mx); sum += p[e]; }
  const float rs = 1.0f / sum;
#pragma unroll
  for (int e = 0; e < E_; e++) p[e] *= rs;

  // top-2 (jax tie-break: lower index wins -> strict >)
  int i1 = 0;
#pragma unroll
  for (int e = 1; e < E_; e++) if (p[e] > p[i1]) i1 = e;
  int i2 = (i1 == 0) ? 1 : 0;
#pragma unroll
  for (int e = 0; e < E_; e++) if (e != i1 && p[e] > p[i2]) i2 = e;

  if (lane < E_)
    cw[(size_t)token * E_ + lane] = (lane == i1 || lane == i2) ? p[lane] : 0.f;
  if (lane == 0) {
    int p1 = atomicAdd(&cnt[i1], 1);
    toklist[i1 * CAP + p1] = token;
    int p2 = atomicAdd(&cnt[i2], 1);
    toklist[i2 * CAP + p2] = token;
  }
}

// ---------------------------------------------------------------------------
// Paired WMMA K-loop: one A fragment feeds two N-tiles (A reuse; 3 loads/WMMA).
// Ap  = lane's A row ptr (LDS, offset by halfsel*16)
// Bp0/Bp1 = lane's B row ptrs for the two N-tiles (global, offset halfsel*16)
// ---------------------------------------------------------------------------
__device__ __forceinline__ void wmma_kloop2(const _Float16* Ap,
                                            const _Float16* __restrict__ Bp0,
                                            const _Float16* __restrict__ Bp1,
                                            int K, v8f& c0, v8f& c1) {
  for (int k0 = 0; k0 < K; k0 += 32) {
    v8h a0 = *(const v8h*)(Ap + k0);
    v8h a1 = *(const v8h*)(Ap + k0 + 8);
    v16h a = __builtin_shufflevector(a0, a1, 0,1,2,3,4,5,6,7,8,9,10,11,12,13,14,15);
    v8h p0 = *(const v8h*)(Bp0 + k0);
    v8h p1 = *(const v8h*)(Bp0 + k0 + 8);
    v16h b0 = __builtin_shufflevector(p0, p1, 0,1,2,3,4,5,6,7,8,9,10,11,12,13,14,15);
    v8h q0 = *(const v8h*)(Bp1 + k0);
    v8h q1 = *(const v8h*)(Bp1 + k0 + 8);
    v16h b1 = __builtin_shufflevector(q0, q1, 0,1,2,3,4,5,6,7,8,9,10,11,12,13,14,15);
    c0 = __builtin_amdgcn_wmma_f32_16x16x32_f16(false, a, false, b0, (short)0,
                                                c0, false, false);
    c1 = __builtin_amdgcn_wmma_f32_16x16x32_f16(false, a, false, b1, (short)0,
                                                c1, false, false);
  }
}

// ---------------------------------------------------------------------------
// Kernel 4: per-expert gathered MLP over 16-token tiles.
// grid.x = e*TILES + tile (expert-major for L2 weight residency), 256 threads.
// ---------------------------------------------------------------------------
__global__ __launch_bounds__(256)
void expert_mlp_kernel(const float* __restrict__ x,
                       const _Float16* __restrict__ W1t, const float* __restrict__ b1,
                       const _Float16* __restrict__ W2t, const float* __restrict__ b2,
                       const _Float16* __restrict__ W3t, const float* __restrict__ b3,
                       const float* __restrict__ cw,
                       const int* __restrict__ cnt,
                       const int* __restrict__ toklist,
                       float* __restrict__ out) {
  extern __shared__ char smem[];
  _Float16* Xb   = (_Float16*)(smem + LDS_XB);
  _Float16* H1   = (_Float16*)(smem + LDS_H1);
  _Float16* H2   = (_Float16*)(smem + LDS_H2);
  float*    rowW = (float*)(smem + LDS_ROWW);
  int*      rTok = (int*)(smem + LDS_RTOK);

  const int e    = blockIdx.x >> 8;        // / TILES
  const int tile = blockIdx.x & (TILES - 1);
  const int n_e  = cnt[e];
  if (tile * TILE_T >= n_e) return;

  const int tid = threadIdx.x;
  if (tid < TILE_T) {
    const int idx = tile * TILE_T + tid;
    const int tok = (idx < n_e) ? toklist[e * CAP + idx] : -1;
    rTok[tid] = tok;
    rowW[tid] = (tok >= 0) ? cw[(size_t)tok * E_ + e] : 0.f;
  }
  __syncthreads();

  // stage raw x tile -> f16 LDS [16][H]; each thread handles 64 columns
  {
    const int r  = tid >> 4;
    const int c0 = (tid & 15) * 64;
    const int tok = rTok[r];
    const float* xr = x + (size_t)(tok >= 0 ? tok : 0) * H_;
#pragma unroll
    for (int j = 0; j < 64; j += 4) {
      float4 v = (tok >= 0) ? *(const float4*)(xr + c0 + j)
                            : make_float4(0.f, 0.f, 0.f, 0.f);
      _Float16* d = Xb + r * H_ + c0 + j;
      d[0] = (_Float16)v.x; d[1] = (_Float16)v.y;
      d[2] = (_Float16)v.z; d[3] = (_Float16)v.w;
    }
  }
  __syncthreads();

  const int wid     = tid >> 5;
  const int lane    = tid & 31;
  const int lrow    = lane & 15;      // A row / B-C column within tile
  const int halfsel = lane >> 4;      // hi/lo lane half
  const int koffA   = halfsel * 16;
  const int koffB   = halfsel * 16;

  const _Float16* W1e = W1t + (size_t)e * M_ * H_;  // [M][H]
  const _Float16* W2e = W2t + (size_t)e * M_ * M_;  // [M][M]
  const _Float16* W3e = W3t + (size_t)e * H_ * M_;  // [H][M]

  // ---- layer 1: H1 = relu(Xb @ W1 + b1), N = M_, K = H_ ----
  {
    const int nbase = wid * (M_ / 8);
    const _Float16* Ap = Xb + lrow * H_ + koffA;
    for (int nt = 0; nt < (M_ / 8) / 16; nt += 2) {
      const int n0 = nbase + nt * 16 + lrow;
      const int n1 = n0 + 16;
      const _Float16* Bp0 = W1e + (size_t)n0 * H_ + koffB;
      const _Float16* Bp1 = W1e + (size_t)n1 * H_ + koffB;
      __builtin_prefetch(Bp1 + H_, 0, 1);       // next pair's first row
      v8f c0 = {}, c1 = {};
      wmma_kloop2(Ap, Bp0, Bp1, H_, c0, c1);
      const float bias0 = b1[e * M_ + n0];
      const float bias1 = b1[e * M_ + n1];
#pragma unroll
      for (int i = 0; i < 8; i++) {
        float v0 = c0[i] + bias0;
        float v1 = c1[i] + bias1;
        v0 = v0 > 0.f ? v0 : 0.f;
        v1 = v1 > 0.f ? v1 : 0.f;
        H1[(halfsel * 8 + i) * M_ + n0] = (_Float16)v0;
        H1[(halfsel * 8 + i) * M_ + n1] = (_Float16)v1;
      }
    }
  }
  __syncthreads();

  // ---- layer 2: H2 = relu(H1 @ W2 + b2), N = M_, K = M_ ----
  {
    const int nbase = wid * (M_ / 8);
    const _Float16* Ap = H1 + lrow * M_ + koffA;
    for (int nt = 0; nt < (M_ / 8) / 16; nt += 2) {
      const int n0 = nbase + nt * 16 + lrow;
      const int n1 = n0 + 16;
      const _Float16* Bp0 = W2e + (size_t)n0 * M_ + koffB;
      const _Float16* Bp1 = W2e + (size_t)n1 * M_ + koffB;
      __builtin_prefetch(Bp1 + M_, 0, 1);
      v8f c0 = {}, c1 = {};
      wmma_kloop2(Ap, Bp0, Bp1, M_, c0, c1);
      const float bias0 = b2[e * M_ + n0];
      const float bias1 = b2[e * M_ + n1];
#pragma unroll
      for (int i = 0; i < 8; i++) {
        float v0 = c0[i] + bias0;
        float v1 = c1[i] + bias1;
        v0 = v0 > 0.f ? v0 : 0.f;
        v1 = v1 > 0.f ? v1 : 0.f;
        H2[(halfsel * 8 + i) * M_ + n0] = (_Float16)v0;
        H2[(halfsel * 8 + i) * M_ + n1] = (_Float16)v1;
      }
    }
  }
  __syncthreads();

  // ---- layer 3: out += cw * (H2 @ W3 + b3), N = H_, K = M_ ----
  {
    const int nbase = wid * (H_ / 8);
    const _Float16* Ap = H2 + lrow * M_ + koffA;
    for (int nt = 0; nt < (H_ / 8) / 16; nt += 2) {
      const int n0 = nbase + nt * 16 + lrow;
      const int n1 = n0 + 16;
      const _Float16* Bp0 = W3e + (size_t)n0 * M_ + koffB;
      const _Float16* Bp1 = W3e + (size_t)n1 * M_ + koffB;
      __builtin_prefetch(Bp1 + M_, 0, 1);
      v8f c0 = {}, c1 = {};
      wmma_kloop2(Ap, Bp0, Bp1, M_, c0, c1);
      const float bias0 = b3[e * H_ + n0];
      const float bias1 = b3[e * H_ + n1];
#pragma unroll
      for (int i = 0; i < 8; i++) {
        const int r   = halfsel * 8 + i;
        const int tok = rTok[r];
        if (tok >= 0) {
          const float w = rowW[r];
          unsafeAtomicAdd(out + (size_t)tok * H_ + n0, (c0[i] + bias0) * w);
          unsafeAtomicAdd(out + (size_t)tok * H_ + n1, (c1[i] + bias1) * w);
        }
      }
    }
  }
}

// ---------------------------------------------------------------------------
extern "C" void kernel_launch(void* const* d_in, const int* in_sizes, int n_in,
                              void* d_out, int out_size, void* d_ws, size_t ws_size,
                              hipStream_t stream) {
  const float* x     = (const float*)d_in[0];
  const float* genre = (const float*)d_in[1];
  const float* rms_w = (const float*)d_in[2];
  const float* wg_W  = (const float*)d_in[3];
  const float* wg_b  = (const float*)d_in[4];
  const float* gg_W  = (const float*)d_in[5];
  const float* gg_b  = (const float*)d_in[6];
  const float* W1    = (const float*)d_in[7];
  const float* b1    = (const float*)d_in[8];
  const float* W2    = (const float*)d_in[9];
  const float* b2    = (const float*)d_in[10];
  const float* W3    = (const float*)d_in[11];
  const float* b3    = (const float*)d_in[12];
  float* out = (float*)d_out;

  char* ws = (char*)d_ws;
  int*      cnt  = (int*)(ws + WS_CNT);
  float*    gg   = (float*)(ws + WS_GG);
  float*    cw   = (float*)(ws + WS_CW);
  int*      tokl = (int*)(ws + WS_TOK);
  _Float16* W1t  = (_Float16*)(ws + WS_W1T);
  _Float16* W2t  = (_Float16*)(ws + WS_W2T);
  _Float16* W3t  = (_Float16*)(ws + WS_W3T);

  // zero output accumulator + dispatch counters
  hipMemsetAsync(out, 0, (size_t)NTOK * H_ * sizeof(float), stream);
  hipMemsetAsync(cnt, 0, E_ * sizeof(int), stream);

  // one-time (per launch) weight transpose + f16 convert
  dim3 tb(32, 8, 1);
  transpose_cvt_kernel<<<dim3(M_ / 32, H_ / 32, E_), tb, 0, stream>>>(W1, W1t, H_, M_);
  transpose_cvt_kernel<<<dim3(M_ / 32, M_ / 32, E_), tb, 0, stream>>>(W2, W2t, M_, M_);
  transpose_cvt_kernel<<<dim3(H_ / 32, M_ / 32, E_), tb, 0, stream>>>(W3, W3t, M_, H_);

  genre_gate_kernel<<<dim3(B_ * E_), dim3(32), 0, stream>>>(genre, gg_W, gg_b, wg_b, gg);

  gate_topk_kernel<<<dim3(NTOK / 4), dim3(128), 0, stream>>>(x, rms_w, wg_W, gg,
                                                             cw, cnt, tokl);

  hipFuncSetAttribute((const void*)expert_mlp_kernel,
                      hipFuncAttributeMaxDynamicSharedMemorySize, SMEM_BYTES);
  expert_mlp_kernel<<<dim3(E_ * TILES), dim3(256), SMEM_BYTES, stream>>>(
      x, W1t, b1, W2t, b2, W3t, b3, cw, cnt, tokl, out);
}
